// GCN_53094385713561
// MI455X (gfx1250) — compile-verified, matrix operand order checked by reading, not run on hardware
//
#include <hip/hip_runtime.h>
#include <hip/hip_bf16.h>

typedef __attribute__((ext_vector_type(16))) _Float16 v16h;
typedef __attribute__((ext_vector_type(8)))  float    v8f;

union U16h { unsigned int u[8]; v16h h; };

#define NODES 50000
#define EDGES 800000
#define HDIM  128
#define CATD  384
#define NGRAPH 512

// ---------------- small elementwise / setup kernels ----------------

__global__ void set_const_f32(float* p, float v, int n) {
    int i = blockIdx.x * blockDim.x + threadIdx.x;
    if (i < n) p[i] = v;
}

__global__ void deg_count(const int* __restrict__ ei, float* __restrict__ deg, int E) {
    int e = blockIdx.x * blockDim.x + threadIdx.x;
    if (e < E) atomicAdd(&deg[ei[E + e]], 1.0f);
}

__global__ void rsqrt_inplace(float* p, int n) {
    int i = blockIdx.x * blockDim.x + threadIdx.x;
    if (i < n) p[i] = rsqrtf(p[i]);   // deg >= 1 always (self-loop)
}

__global__ void edge_norm(const int* __restrict__ ei, const float* __restrict__ dinv,
                          float* __restrict__ nrm, int E) {
    int e = blockIdx.x * blockDim.x + threadIdx.x;
    if (e < E) nrm[e] = dinv[ei[e]] * dinv[ei[E + e]];
}

__global__ void cvt_f32_to_f16(const float* __restrict__ x, _Float16* __restrict__ y, int n) {
    int i = blockIdx.x * blockDim.x + threadIdx.x;
    if (i < n) y[i] = (_Float16)x[i];
}

// Wt16[n*K + k] = (f16) W[k*Nc + n]  (transpose so GEMM B loads are packed pairs)
__global__ void cvt_w_transpose(const float* __restrict__ W, _Float16* __restrict__ Wt,
                                int K, int Nc) {
    int i = blockIdx.x * blockDim.x + threadIdx.x;
    if (i >= K * Nc) return;
    int k = i / Nc, n = i - k * Nc;
    Wt[n * K + k] = (_Float16)W[k * Nc + n];
}

// agg[i][f] = bias[f] + t[i][f] * dinv[i]^2     (self-loop message + bias)
__global__ void init_agg(const float* __restrict__ t, const float* __restrict__ bias,
                         const float* __restrict__ dinv, float* __restrict__ agg, int total) {
    int idx = blockIdx.x * blockDim.x + threadIdx.x;
    if (idx >= total) return;
    int i = idx >> 7, f = idx & 127;
    float d = dinv[i];
    agg[idx] = bias[f] + t[idx] * d * d;
}

// one thread per (edge, feature): agg[dst][f] += t[src][f] * norm[e]
__global__ void edge_scatter(const float* __restrict__ t, const int* __restrict__ ei,
                             const float* __restrict__ nrm, float* __restrict__ agg, int E) {
    int idx = blockIdx.x * blockDim.x + threadIdx.x;
    if (idx >= E * HDIM) return;
    int e = idx >> 7, f = idx & 127;
    int s = ei[e], d = ei[E + e];
    atomicAdd(&agg[d * HDIM + f], t[s * HDIM + f] * nrm[e]);
}

// ELU then write f16 into concat buffer column block [layer*128, layer*128+128)
__global__ void elu_to_cat(const float* __restrict__ agg, _Float16* __restrict__ hcat16,
                           int layer, int total) {
    int idx = blockIdx.x * blockDim.x + threadIdx.x;
    if (idx >= total) return;
    int i = idx >> 7, f = idx & 127;
    float v = agg[idx];
    v = v > 0.0f ? v : (expf(v) - 1.0f);
    hcat16[i * CATD + layer * HDIM + f] = (_Float16)v;
}

// y = BN(leaky_relu(x + bias)); optional f32 and/or f16 outputs
__global__ void bias_lrelu_bn(const float* __restrict__ x, const float* __restrict__ b,
                              const float* __restrict__ g, const float* __restrict__ be,
                              const float* __restrict__ m, const float* __restrict__ v,
                              float* __restrict__ y32, _Float16* __restrict__ y16, int total) {
    int idx = blockIdx.x * blockDim.x + threadIdx.x;
    if (idx >= total) return;
    int f = idx & 127;
    float t = x[idx] + b[f];
    t = t > 0.0f ? t : 0.01f * t;
    t = (t - m[f]) * rsqrtf(v[f] + 1e-5f) * g[f] + be[f];
    if (y32) y32[idx] = t;
    if (y16) y16[idx] = (_Float16)t;
}

__global__ void pool_scatter(const float* __restrict__ h, const int* __restrict__ batch,
                             float* __restrict__ gsum, float* __restrict__ gcnt, int total) {
    int idx = blockIdx.x * blockDim.x + threadIdx.x;
    if (idx >= total) return;
    int i = idx >> 7, f = idx & 127;
    int g = batch[i];
    atomicAdd(&gsum[g * HDIM + f], h[idx]);
    if (f == 0) atomicAdd(&gcnt[g], 1.0f);
}

__global__ void pool_finish(const float* __restrict__ gsum, const float* __restrict__ gcnt,
                            _Float16* __restrict__ pooled16, int total) {
    int idx = blockIdx.x * blockDim.x + threadIdx.x;
    if (idx >= total) return;
    int g = idx >> 7;
    pooled16[idx] = (_Float16)(gsum[idx] / fmaxf(gcnt[g], 1.0f));
}

// ---------------- WMMA GEMM:  C[M x 128] = A[M x K] @ B[K x 128] ----------------
// A: f16 row-major with row stride lda; Bt: f16, B transposed -> Bt[n*K + k].
// Block = 256 threads = 8 waves; wave w owns the 16-col tile w; block b owns rows [16b,16b+16).
// Emits v_wmma_f32_16x16x32_f16; K must be a multiple of 32; EXEC all-ones (no divergence).
__global__ void wmma_gemm_f16(const _Float16* __restrict__ A, int lda,
                              const _Float16* __restrict__ Bt,
                              float* __restrict__ C, int K) {
    const int wave = threadIdx.x >> 5;
    const int lane = threadIdx.x & 31;
    const int row0 = blockIdx.x * 16;
    const int col0 = wave * 16;
    const int mr   = lane & 15;
    const int hs   = (lane >> 4) ? 8 : 0;           // K offset for high half-wave

    const _Float16* arow = A  + (long)(row0 + mr) * lda;
    const _Float16* brow = Bt + (long)(col0 + mr) * K;

    v8f acc = {};
    for (int kk = 0; kk < K; kk += 32) {
        U16h a, b;
#pragma unroll
        for (int v = 0; v < 8; ++v) {
            // 16-bit A-matrix 16x32 layout: VGPR v holds K = 2(v&3) + 8*(lane>=16) + 16*(v>=4)
            int kb = kk + ((v & 3) << 1) + hs + ((v >> 2) << 4);
            a.u[v] = *(const unsigned int*)(arow + kb);   // packed pair (k, k+1)
            b.u[v] = *(const unsigned int*)(brow + kb);
        }
        acc = __builtin_amdgcn_wmma_f32_16x16x32_f16(
                  false, a.h, false, b.h, (short)0, acc, false, false);
    }

    // C/D layout: lane -> column (lane&15); VGPR r -> row r (lanes 0-15) / r+8 (lanes 16-31)
    const int n     = col0 + (lane & 15);
    const int mbase = row0 + ((lane >> 4) ? 8 : 0);
#pragma unroll
    for (int r = 0; r < 8; ++r)
        C[(long)(mbase + r) * HDIM + n] = acc[r];
}

// ---------------- final head: 128 -> 16 logits + log_softmax, one wave per graph ----------------
__global__ void final_head(const float* __restrict__ hb, const float* __restrict__ Wc,
                           const float* __restrict__ bc, float* __restrict__ out) {
    int g = blockIdx.x;
    int lane = threadIdx.x;          // 32 lanes; lanes 16..31 mirror 0..15
    int j = lane & 15;
    float acc = bc[j];
    const float* hr = hb + g * HDIM;
#pragma unroll 4
    for (int k = 0; k < HDIM; ++k) acc += hr[k] * Wc[k * 16 + j];
    float mx = acc;
    for (int s = 1; s < 16; s <<= 1) mx = fmaxf(mx, __shfl_xor(mx, s));
    float e = expf(acc - mx);
    float se = e;
    for (int s = 1; s < 16; s <<= 1) se += __shfl_xor(se, s);
    if (lane < 16) out[g * 16 + j] = acc - mx - logf(se);
}

// ---------------- host orchestration ----------------

extern "C" void kernel_launch(void* const* d_in, const int* in_sizes, int n_in,
                              void* d_out, int out_size, void* d_ws, size_t ws_size,
                              hipStream_t stream) {
    (void)in_sizes; (void)n_in; (void)out_size; (void)ws_size;

    const float* x     = (const float*)d_in[0];
    const int*   ei    = (const int*)d_in[1];
    const int*   batch = (const int*)d_in[2];
    const float* convW[3] = {(const float*)d_in[3], (const float*)d_in[4], (const float*)d_in[5]};
    const float* convB[3] = {(const float*)d_in[6], (const float*)d_in[7], (const float*)d_in[8]};
    const float *W1 = (const float*)d_in[9],  *b1 = (const float*)d_in[10],
                *g1 = (const float*)d_in[11], *be1 = (const float*)d_in[12],
                *m1 = (const float*)d_in[13], *v1 = (const float*)d_in[14],
                *W2 = (const float*)d_in[15], *b2 = (const float*)d_in[16],
                *g2 = (const float*)d_in[17], *be2 = (const float*)d_in[18],
                *m2 = (const float*)d_in[19], *v2 = (const float*)d_in[20];
    const float *Wa = (const float*)d_in[21], *ba = (const float*)d_in[22],
                *ga = (const float*)d_in[23], *bea = (const float*)d_in[24],
                *ma = (const float*)d_in[25], *va = (const float*)d_in[26],
                *Wb = (const float*)d_in[27], *bb = (const float*)d_in[28],
                *gb = (const float*)d_in[29], *beb = (const float*)d_in[30],
                *mb = (const float*)d_in[31], *vb = (const float*)d_in[32],
                *Wc = (const float*)d_in[33], *bc = (const float*)d_in[34];

    // ----- workspace carve-out (all 256B aligned) -----
    char* base = (char*)d_ws; size_t off = 0;
    auto take = [&](size_t bytes) -> void* {
        void* r = base + off;
        off += (bytes + 255) & ~(size_t)255;
        return r;
    };
    float*     dinv     = (float*)take((size_t)NODES * 4);
    float*     nrm      = (float*)take((size_t)EDGES * 4);
    _Float16*  act16    = (_Float16*)take((size_t)NODES * HDIM * 2);   // x16 / proj1-out reuse
    _Float16*  hcat16   = (_Float16*)take((size_t)NODES * CATD * 2);
    float*     tbuf     = (float*)take((size_t)NODES * HDIM * 4);      // GEMM outputs
    float*     agg      = (float*)take((size_t)NODES * HDIM * 4);      // aggregation / 2nd GEMM out
    _Float16*  wt16     = (_Float16*)take((size_t)CATD * HDIM * 2);    // transposed f16 weights
    float*     gsum     = (float*)take((size_t)NGRAPH * HDIM * 4);
    float*     gcnt     = (float*)take((size_t)NGRAPH * 4);
    _Float16*  pooled16 = (_Float16*)take((size_t)NGRAPH * HDIM * 2);
    _Float16*  mact16   = (_Float16*)take((size_t)NGRAPH * HDIM * 2);

    const int TB = 256;
    auto G = [&](long n) { return dim3((unsigned)((n + TB - 1) / TB)); };
    const long NH = (long)NODES * HDIM;
    const long EH = (long)EDGES * HDIM;
    const long PH = (long)NGRAPH * HDIM;

    // ----- gcn_norm: deg (self-loop folded into init=1), dinv, per-edge norm -----
    set_const_f32<<<G(NODES), TB, 0, stream>>>(dinv, 1.0f, NODES);
    deg_count<<<G(EDGES), TB, 0, stream>>>(ei, dinv, EDGES);
    rsqrt_inplace<<<G(NODES), TB, 0, stream>>>(dinv, NODES);
    edge_norm<<<G(EDGES), TB, 0, stream>>>(ei, dinv, nrm, EDGES);

    // ----- input features to f16 -----
    cvt_f32_to_f16<<<G(NH), TB, 0, stream>>>(x, act16, (int)NH);

    // ----- 3 GCN conv layers -----
    for (int l = 0; l < 3; ++l) {
        cvt_w_transpose<<<G(HDIM * HDIM), TB, 0, stream>>>(convW[l], wt16, HDIM, HDIM);
        const _Float16* A = (l == 0) ? act16 : (hcat16 + (l - 1) * HDIM);
        int lda = (l == 0) ? HDIM : CATD;
        wmma_gemm_f16<<<dim3(NODES / 16), 256, 0, stream>>>(A, lda, wt16, tbuf, HDIM);
        init_agg<<<G(NH), TB, 0, stream>>>(tbuf, convB[l], dinv, agg, (int)NH);
        edge_scatter<<<G(EH), TB, 0, stream>>>(tbuf, ei, nrm, agg, EDGES);
        elu_to_cat<<<G(NH), TB, 0, stream>>>(agg, hcat16, l, (int)NH);
    }

    // ----- projection: [N,384] @ W1 -> BN -> [N,128] @ W2 -> BN -----
    cvt_w_transpose<<<G(CATD * HDIM), TB, 0, stream>>>(W1, wt16, CATD, HDIM);
    wmma_gemm_f16<<<dim3(NODES / 16), 256, 0, stream>>>(hcat16, CATD, wt16, tbuf, CATD);
    bias_lrelu_bn<<<G(NH), TB, 0, stream>>>(tbuf, b1, g1, be1, m1, v1,
                                            nullptr, act16, (int)NH);
    cvt_w_transpose<<<G(HDIM * HDIM), TB, 0, stream>>>(W2, wt16, HDIM, HDIM);
    wmma_gemm_f16<<<dim3(NODES / 16), 256, 0, stream>>>(act16, HDIM, wt16, agg, HDIM);
    bias_lrelu_bn<<<G(NH), TB, 0, stream>>>(agg, b2, g2, be2, m2, v2,
                                            tbuf, nullptr, (int)NH);

    // ----- global mean pool -----
    set_const_f32<<<G(PH), TB, 0, stream>>>(gsum, 0.0f, (int)PH);
    set_const_f32<<<G(NGRAPH), TB, 0, stream>>>(gcnt, 0.0f, NGRAPH);
    pool_scatter<<<G(NH), TB, 0, stream>>>(tbuf, batch, gsum, gcnt, (int)NH);
    pool_finish<<<G(PH), TB, 0, stream>>>(gsum, gcnt, pooled16, (int)PH);

    // ----- MLP head (M = 512 = 32 row tiles) -----
    cvt_w_transpose<<<G(HDIM * HDIM), TB, 0, stream>>>(Wa, wt16, HDIM, HDIM);
    wmma_gemm_f16<<<dim3(NGRAPH / 16), 256, 0, stream>>>(pooled16, HDIM, wt16, tbuf, HDIM);
    bias_lrelu_bn<<<G(PH), TB, 0, stream>>>(tbuf, ba, ga, bea, ma, va,
                                            nullptr, mact16, (int)PH);
    cvt_w_transpose<<<G(HDIM * HDIM), TB, 0, stream>>>(Wb, wt16, HDIM, HDIM);
    wmma_gemm_f16<<<dim3(NGRAPH / 16), 256, 0, stream>>>(mact16, HDIM, wt16, agg, HDIM);
    bias_lrelu_bn<<<G(PH), TB, 0, stream>>>(agg, bb, gb, beb, mb, vb,
                                            gsum, nullptr, (int)PH);   // reuse gsum as f32 hidden

    // ----- final linear 128->16 + log_softmax, one wave per graph -----
    final_head<<<dim3(NGRAPH), 32, 0, stream>>>(gsum, Wc, bc, (float*)d_out);
}
